// LinearCatVAE_78125455114285
// MI455X (gfx1250) — matrix operand order repaired and checked by reading, not compiled, not run on hardware
//
#include <hip/hip_runtime.h>
#include <hip/hip_bf16.h>
#include <math.h>

// ---------------------------------------------------------------------------
// LinearCatVAE loss, MI455X (gfx1250, wave32).
// B=8192, D=2000, N=D-1=1999 (padded to 2048), K=64.
// Helmert structure of Psi -> suffix/prefix scans replace the two B x D x D
// GEMMs. The three K=64 GEMMs use v_wmma_f32_16x16x32_f16.
// ---------------------------------------------------------------------------

#define BB 8192
#define DD 2000
#define NN 1999
#define NP 2048
#define KK 64

typedef __attribute__((ext_vector_type(16))) _Float16 v16h;
typedef __attribute__((ext_vector_type(8)))  float    v8f;
typedef __attribute__((ext_vector_type(8)))  _Float16 h8v;

union V16 { v16h v; h8v h8[2]; };

// ---- workspace layout (bytes) ----
#define OFF_ACC    0          // double[4]: 0=mult, 1=logit_quad, 2=prior, 3=logdet
#define OFF_A      256        // float[2048]  Psi diag
#define OFF_C      8448       // float[2048]  Psi off-diag
#define OFF_MROW   16640      // float[8192]  gammaln(sum x+1)-sum gammaln(x+1)
#define OFF_XS     49408      // float[8192]  row sums of x
#define OFF_D2     82176      // float[8192]  row sums of diff^2
#define OFF_WG     114944     // float[64*64] Wdec^T Wdec
#define OFF_MINV   131328     // float[64*64] M^{-1}
#define OFF_WENCH  147712     // f16[64*2048]   W_enc padded
#define OFF_WDECH  409856     // f16[2048*64]   W_dec padded
#define OFF_WDECTH 672000     // f16[64*2048]   W_dec^T padded
#define OFF_ZMH    934144     // f16[8192*64]   z_mean
#define OFF_T      1982720    // float[8192*64] t = diff @ W_dec
#define OFF_HXH    4079872    // f16[8192*2048] hx
#define OFF_DIFFH  37634304   // f16[8192*2048] diff

__device__ inline float wave_sum(float v) {
  for (int o = 16; o; o >>= 1) v += __shfl_xor(v, o, 32);
  return v;
}
__device__ inline float wave_max(float v) {
  for (int o = 16; o; o >>= 1) v = fmaxf(v, __shfl_xor(v, o, 32));
  return v;
}

// A fragment: 16x32 f16 tile, row-major source, ISA 16-bit A layout.
__device__ inline v16h load_afrag(const _Float16* __restrict__ base, int lda,
                                  int r0, int k0, int lane) {
  int r  = lane & 15;
  int hk = (lane >> 4) * 8;
  const _Float16* p = base + (size_t)(r0 + r) * lda + k0 + hk;
  V16 u;
  u.h8[0] = *(const h8v*)(p);        // K = hk .. hk+7      (+0  or +8)
  u.h8[1] = *(const h8v*)(p + 16);   // K = 16+hk .. 16+hk+7
  return u.v;
}

// B fragment: 32x16 f16 tile; source supplied TRANSPOSED as Bt[n][k] row-major,
// so each lane reads two runs of 8 contiguous halfs, 16 apart. ISA 16-bit B layout.
__device__ inline v16h load_bfrag(const _Float16* __restrict__ baseT, int ldb,
                                  int n0, int k0, int lane) {
  int n  = lane & 15;
  int ko = (lane >> 4) * 8;
  const _Float16* p = baseT + (size_t)(n0 + n) * ldb + k0 + ko;
  V16 u;
  u.h8[0] = *(const h8v*)(p);
  u.h8[1] = *(const h8v*)(p + 16);
  return u.v;
}

// ---------------------------------------------------------------------------
// K0: zero accumulators/d2, extract Helmert diag/offdiag, build f16 weights.
__global__ void k_prep(const float* __restrict__ Psi,
                       const float* __restrict__ Wenc,
                       const float* __restrict__ Wdec,
                       double* __restrict__ acc,
                       float* __restrict__ a, float* __restrict__ c,
                       float* __restrict__ d2,
                       _Float16* __restrict__ wench,
                       _Float16* __restrict__ wdech,
                       _Float16* __restrict__ wdecth) {
  int idx = blockIdx.x * 256 + threadIdx.x;
  if (idx < 4)    acc[idx] = 0.0;
  if (idx < BB)   d2[idx] = 0.f;
  if (idx < NP) {
    a[idx] = (idx < NN) ? Psi[(size_t)idx * DD + idx] : 0.f;
    c[idx] = (idx < NN) ? Psi[(size_t)idx * DD + idx + 1] : 0.f;
  }
  if (idx < KK * NP) {                      // 64 x 2048
    int r = idx >> 11, kk = idx & (NP - 1);
    wench[idx]  = (kk < NN) ? (_Float16)Wenc[(size_t)r * NN + kk] : (_Float16)0.f;
    wdecth[idx] = (kk < NN) ? (_Float16)Wdec[(size_t)kk * KK + r] : (_Float16)0.f;
  }
  if (idx < NP * KK) {                      // 2048 x 64
    int d = idx >> 6, k = idx & 63;
    wdech[idx] = (d < NN) ? (_Float16)Wdec[(size_t)d * KK + k] : (_Float16)0.f;
  }
}

// K1: Wg = Wdec^T Wdec (64x64, fp32, exact inputs).
__global__ void k_wg(const float* __restrict__ Wdec, float* __restrict__ Wg) {
  int idx = blockIdx.x * 256 + threadIdx.x;
  if (idx >= KK * KK) return;
  int i = idx >> 6, j = idx & 63;
  float s = 0.f;
  for (int d = 0; d < NN; ++d)
    s += Wdec[(size_t)d * KK + i] * Wdec[(size_t)d * KK + j];
  Wg[idx] = s;
}

// K2: M = diag(exp(-vlv)) + Wg/var; Cholesky; logdet; explicit M^{-1}.
__global__ void k_chol(const float* __restrict__ Wg,
                       const float* __restrict__ vlv,
                       const float* __restrict__ lsq,
                       float* __restrict__ Minv, double* __restrict__ acc) {
  __shared__ float Lm[KK * KK];
  int t = threadIdx.x;  // 64 threads
  float var = expf(lsq[0]);
  for (int idx = t; idx < KK * KK; idx += KK) {
    int i = idx >> 6, j = idx & 63;
    float m = Wg[idx] / var;
    if (i == j) m += expf(-vlv[i]);
    Lm[idx] = m;
  }
  __syncthreads();
  for (int j = 0; j < KK; ++j) {
    if (t == j) {
      float s = Lm[j * KK + j];
      for (int p = 0; p < j; ++p) s -= Lm[j * KK + p] * Lm[j * KK + p];
      Lm[j * KK + j] = sqrtf(s);
    }
    __syncthreads();
    if (t > j) {
      float s = Lm[t * KK + j];
      for (int p = 0; p < j; ++p) s -= Lm[t * KK + p] * Lm[j * KK + p];
      Lm[t * KK + j] = s / Lm[j * KK + j];
    }
    __syncthreads();
  }
  if (t == 0) {
    double ld = (double)NN * (double)lsq[0];
    for (int k = 0; k < KK; ++k) ld += (double)vlv[k];
    for (int j = 0; j < KK; ++j) ld += 2.0 * log((double)Lm[j * KK + j]);
    acc[3] = ld;
  }
  __syncthreads();
  // column t of M^{-1}: forward then backward substitution.
  float z[KK], w[KK];
  for (int k = 0; k < KK; ++k) {
    float s = (k == t) ? 1.f : 0.f;
    for (int p = 0; p < k; ++p) s -= Lm[k * KK + p] * z[p];
    z[k] = s / Lm[k * KK + k];
  }
  for (int k = KK - 1; k >= 0; --k) {
    float s = z[k];
    for (int p = k + 1; p < KK; ++p) s -= Lm[p * KK + k] * w[p];
    w[k] = s / Lm[k * KK + k];
  }
  for (int k = 0; k < KK; ++k) Minv[k * KK + t] = w[k];
}

// K3: per-row (one wave/row): L=log(x+1), suffix scan -> hx (f16, padded),
// plus multinomial constants gammaln(sum+1)-sum gammaln(x+1) and sum(x).
__global__ void k_hx(const float* __restrict__ x,
                     const float* __restrict__ a, const float* __restrict__ c,
                     _Float16* __restrict__ hxh,
                     float* __restrict__ mrow, float* __restrict__ xsb) {
  int wid  = (blockIdx.x * blockDim.x + threadIdx.x) >> 5;
  int lane = threadIdx.x & 31;
  if (wid >= BB) return;
  const float* xr = x + (size_t)wid * DD;
  _Float16*    hr = hxh + (size_t)wid * NP;
  float carry = 0.f, xs = 0.f, gs = 0.f;
  for (int ch = 63; ch >= 0; --ch) {           // descending chunks of 32
    int j = ch * 32 + lane;                    // j in [0,2048)
    float xv = (j < DD) ? xr[j] : 0.f;
    float v  = (j < DD) ? logf(xv + 1.f) : 0.f;
    if (j < DD) { xs += xv; gs += lgammaf(xv + 1.f); }
    float s = v;                               // in-wave suffix-inclusive scan
    for (int off = 1; off < 32; off <<= 1) {
      float nv = __shfl_down(s, off, 32);
      if (lane + off < 32) s += nv;
    }
    float S = s + carry;                       // suffix sum starting at j
    float hxv = a[j] * v + c[j] * (S - v);     // a_j*L_j + c_j*SuffixSum_{>j}
    hr[j] = (j < NN) ? (_Float16)hxv : (_Float16)0.f;
    carry += __shfl(s, 0, 32);                 // total of this chunk
  }
  xs = wave_sum(xs);
  gs = wave_sum(gs);
  if (lane == 0) {
    mrow[wid] = lgammaf(xs + 1.f) - gs;
    xsb[wid]  = xs;
  }
}

// K4: per-row prefix scan -> logits on the fly; two passes for log-softmax;
// accumulates multinomial log-likelihood mean.
__global__ void k_logits(const float* __restrict__ x,
                         const float* __restrict__ eta,
                         const float* __restrict__ a, const float* __restrict__ c,
                         const float* __restrict__ mrow,
                         const float* __restrict__ xsb,
                         double* __restrict__ acc) {
  int wid  = (blockIdx.x * blockDim.x + threadIdx.x) >> 5;
  int lane = threadIdx.x & 31;
  if (wid >= BB) return;
  const float* er = eta + (size_t)wid * NN;
  const float* xr = x + (size_t)wid * DD;
  float carry = 0.f, mmax = -1e30f, dot = 0.f;
  for (int ch = 0; ch < 63; ++ch) {            // ascending chunks; d<=2015
    int d = ch * 32 + lane;
    float ev = (d < NN) ? er[d] : 0.f;
    float g  = c[d] * ev;                      // c padded 0 beyond 1999
    float s  = g;                              // in-wave inclusive prefix scan
    for (int off = 1; off < 32; off <<= 1) {
      float nv = __shfl_up(s, off, 32);
      if (lane >= off) s += nv;
    }
    float logit = carry + (s - g) + ((d < NN) ? a[d] * ev : 0.f);
    if (d < DD) { mmax = fmaxf(mmax, logit); dot += xr[d] * logit; }
    carry += __shfl(s, 31, 32);
  }
  mmax = wave_max(mmax);
  dot  = wave_sum(dot);
  carry = 0.f;
  float es = 0.f;
  for (int ch = 0; ch < 63; ++ch) {            // pass 2: sum exp(logit - max)
    int d = ch * 32 + lane;
    float ev = (d < NN) ? er[d] : 0.f;
    float g  = c[d] * ev;
    float s  = g;
    for (int off = 1; off < 32; off <<= 1) {
      float nv = __shfl_up(s, off, 32);
      if (lane >= off) s += nv;
    }
    float logit = carry + (s - g) + ((d < NN) ? a[d] * ev : 0.f);
    if (d < DD) es += expf(logit - mmax);
    carry += __shfl(s, 31, 32);
  }
  es = wave_sum(es);
  if (lane == 0) {
    double ll = (double)mrow[wid] + (double)dot
              - (double)xsb[wid] * ((double)mmax + (double)logf(es));
    atomicAdd(&acc[0], ll / (double)BB);
  }
}

// K5: z_mean = hx @ Wenc^T via WMMA; writes z_mean (f16) + prior accumulation.
__global__ void k_zmean(const _Float16* __restrict__ hxh,
                        const _Float16* __restrict__ wench,
                        _Float16* __restrict__ zmh, double* __restrict__ acc) {
  int wave = threadIdx.x >> 5, lane = threadIdx.x & 31;
  int m0 = blockIdx.x * 16, n0 = wave * 16;    // 512 blocks x 4 waves
  v8f cacc = {};
  for (int k0 = 0; k0 < NP; k0 += 32) {
    v16h af = load_afrag(hxh,   NP, m0, k0, lane);
    v16h bf = load_bfrag(wench, NP, n0, k0, lane);
    cacc = __builtin_amdgcn_wmma_f32_16x16x32_f16(false, af, false, bf,
                                                  (short)0, cacc, false, false);
  }
  int n = lane & 15, mb = (lane >> 4) * 8;
  float ss = 0.f;
#pragma unroll
  for (int r = 0; r < 8; ++r) {
    float v = cacc[r];
    zmh[(size_t)(m0 + mb + r) * KK + n0 + n] = (_Float16)v;
    ss += v * v;
  }
  ss = wave_sum(ss);
  if (lane == 0) atomicAdd(&acc[2], -0.5 * (double)ss / (double)BB);
}

// K6: mu = z_mean @ Wdec^T (WMMA), fused diff = eta - mu (f16) and per-row
// sum(diff^2) accumulation.
__global__ void k_diff(const _Float16* __restrict__ zmh,
                       const _Float16* __restrict__ wdech,
                       const float* __restrict__ eta,
                       _Float16* __restrict__ diffh, float* __restrict__ d2) {
  int wave = threadIdx.x >> 5, lane = threadIdx.x & 31;
  int tile = blockIdx.x * 4 + wave;            // 16384 blocks x 4 waves
  int m0 = (tile >> 7) * 16, n0 = (tile & 127) * 16;
  v8f cacc = {};
  for (int k0 = 0; k0 < KK; k0 += 32) {
    v16h af = load_afrag(zmh,   KK, m0, k0, lane);
    v16h bf = load_bfrag(wdech, KK, n0, k0, lane);
    cacc = __builtin_amdgcn_wmma_f32_16x16x32_f16(false, af, false, bf,
                                                  (short)0, cacc, false, false);
  }
  int n = lane & 15, mb = (lane >> 4) * 8;
  int d = n0 + n;
  float srow[8];
#pragma unroll
  for (int r = 0; r < 8; ++r) {
    int row = m0 + mb + r;
    float ev = (d < NN) ? eta[(size_t)row * NN + d] : 0.f;
    float dv = (d < NN) ? (ev - cacc[r]) : 0.f;
    diffh[(size_t)row * NP + d] = (_Float16)dv;
    srow[r] = dv * dv;
  }
#pragma unroll
  for (int r = 0; r < 8; ++r)
    for (int off = 1; off < 16; off <<= 1)
      srow[r] += __shfl_xor(srow[r], off, 16);  // reduce over 16-lane half
  if ((lane & 15) == 0) {
#pragma unroll
    for (int r = 0; r < 8; ++r) atomicAdd(&d2[m0 + mb + r], srow[r]);
  }
}

// K7: t = diff @ Wdec via WMMA (B supplied as WdecT).
__global__ void k_tgemm(const _Float16* __restrict__ diffh,
                        const _Float16* __restrict__ wdecth,
                        float* __restrict__ tmat) {
  int wave = threadIdx.x >> 5, lane = threadIdx.x & 31;
  int m0 = blockIdx.x * 16, n0 = wave * 16;    // 512 blocks x 4 waves
  v8f cacc = {};
  for (int k0 = 0; k0 < NP; k0 += 32) {
    v16h af = load_afrag(diffh,  NP, m0, k0, lane);
    v16h bf = load_bfrag(wdecth, NP, n0, k0, lane);
    cacc = __builtin_amdgcn_wmma_f32_16x16x32_f16(false, af, false, bf,
                                                  (short)0, cacc, false, false);
  }
  int n = lane & 15, mb = (lane >> 4) * 8;
#pragma unroll
  for (int r = 0; r < 8; ++r)
    tmat[(size_t)(m0 + mb + r) * KK + n0 + n] = cacc[r];
}

// K8: quad_b = d2_b/var - t_b^T M^{-1} t_b / var^2; accumulate -0.5*quad/B.
__global__ void k_quad(const float* __restrict__ tmat,
                       const float* __restrict__ d2,
                       const float* __restrict__ Minv,
                       const float* __restrict__ lsq, double* __restrict__ acc) {
  __shared__ float Mi[KK * KK];
  for (int idx = threadIdx.x; idx < KK * KK; idx += 256) Mi[idx] = Minv[idx];
  __syncthreads();
  int b = blockIdx.x * 256 + threadIdx.x;      // 32 blocks
  float tr[KK];
  const float* trow = tmat + (size_t)b * KK;
  for (int k = 0; k < KK; ++k) tr[k] = trow[k];
  float q2 = 0.f;
  for (int j = 0; j < KK; ++j) {
    float inner = 0.f;
    for (int k = 0; k < KK; ++k) inner += Mi[j * KK + k] * tr[k];
    q2 += tr[j] * inner;
  }
  float var = expf(lsq[0]);
  double quad = (double)d2[b] / (double)var - (double)q2 / ((double)var * (double)var);
  atomicAdd(&acc[1], -0.5 * quad / (double)BB);
}

// K9: combine pieces -> scalar loss.
__global__ void k_fin(const double* __restrict__ acc, float* __restrict__ out) {
  if (threadIdx.x == 0 && blockIdx.x == 0) {
    const double log2pi = 1.8378770664093454836;
    double logit_loss = acc[1] - 0.5 * ((double)NN * log2pi + acc[3]);
    double prior_loss = acc[2] - 0.5 * (double)KK * log2pi;
    out[0] = (float)(-(acc[0] + logit_loss + prior_loss));
  }
}

// ---------------------------------------------------------------------------
extern "C" void kernel_launch(void* const* d_in, const int* in_sizes, int n_in,
                              void* d_out, int out_size, void* d_ws, size_t ws_size,
                              hipStream_t stream) {
  const float* x    = (const float*)d_in[0];   // (B, D)
  const float* Psi  = (const float*)d_in[1];   // (N, D)
  const float* Wenc = (const float*)d_in[2];   // (K, N)
  const float* Wdec = (const float*)d_in[3];   // (N, K)
  const float* vlv  = (const float*)d_in[4];   // (K,)
  const float* lsq  = (const float*)d_in[5];   // scalar
  const float* eta  = (const float*)d_in[6];   // (B, N)
  float* out = (float*)d_out;

  char* ws = (char*)d_ws;
  double*   acc    = (double*)(ws + OFF_ACC);
  float*    a      = (float*)(ws + OFF_A);
  float*    c      = (float*)(ws + OFF_C);
  float*    mrow   = (float*)(ws + OFF_MROW);
  float*    xsb    = (float*)(ws + OFF_XS);
  float*    d2     = (float*)(ws + OFF_D2);
  float*    Wg     = (float*)(ws + OFF_WG);
  float*    Minv   = (float*)(ws + OFF_MINV);
  _Float16* wench  = (_Float16*)(ws + OFF_WENCH);
  _Float16* wdech  = (_Float16*)(ws + OFF_WDECH);
  _Float16* wdecth = (_Float16*)(ws + OFF_WDECTH);
  _Float16* zmh    = (_Float16*)(ws + OFF_ZMH);
  float*    tmat   = (float*)(ws + OFF_T);
  _Float16* hxh    = (_Float16*)(ws + OFF_HXH);
  _Float16* diffh  = (_Float16*)(ws + OFF_DIFFH);

  k_prep  <<<1024, 256, 0, stream>>>(Psi, Wenc, Wdec, acc, a, c, d2,
                                     wench, wdech, wdecth);
  k_wg    <<<16,   256, 0, stream>>>(Wdec, Wg);
  k_chol  <<<1,     64, 0, stream>>>(Wg, vlv, lsq, Minv, acc);
  k_hx    <<<1024, 256, 0, stream>>>(x, a, c, hxh, mrow, xsb);
  k_logits<<<1024, 256, 0, stream>>>(x, eta, a, c, mrow, xsb, acc);
  k_zmean <<<512,  128, 0, stream>>>(hxh, wench, zmh, acc);
  k_diff  <<<16384,128, 0, stream>>>(zmh, wdech, eta, diffh, d2);
  k_tgemm <<<512,  128, 0, stream>>>(diffh, wdecth, tmat);
  k_quad  <<<32,   256, 0, stream>>>(tmat, d2, Minv, lsq, acc);
  k_fin   <<<1,      1, 0, stream>>>(acc, out);
}
